// VectorQuantizerEMA_19104014532926
// MI455X (gfx1250) — matrix operand order repaired and checked by reading, not compiled, not run on hardware
//
#include <hip/hip_runtime.h>

typedef __attribute__((ext_vector_type(16))) __bf16          v16bf;
typedef __attribute__((ext_vector_type(8)))  float           v8f;
typedef __attribute__((ext_vector_type(16))) unsigned short  v16u;

#define KCODES 512
#define CDIM   128
#define BATCH  32
#define TLEN   4096
#define NTOK   (BATCH * TLEN)          // 131072
#define DECAYF 0.99f
#define ONEMD  0.01f
#define EPSF   1e-5f

// ---- output layout (flat, reference tuple order, float32) ----
#define O_ZQ     ((size_t)0)
#define O_CODES  ((size_t)16777216)          // 32*128*4096
#define O_LOSS   ((size_t)16908288)          // + 131072
#define O_CB     ((size_t)16908289)          // + 1
#define O_ECS    ((size_t)16973825)          // + 65536
#define O_EEMB   ((size_t)16974337)          // + 512

// ---- workspace layout (bytes) ----
#define W_E2     ((size_t)0)                  // 512 f32
#define W_CB16   ((size_t)2048)               // 512*128 u16 (bf16)
#define W_CS     ((size_t)133120)             // 512 f32  (zeroed)
#define W_ES     ((size_t)135168)             // 512*128 f32 (zeroed)
#define W_LOSS   ((size_t)397312)             // 1 f32    (zeroed)
#define W_ZERON  (512 + 65536 + 1)            // contiguous zero region (floats)

__device__ __forceinline__ unsigned short f2bf(float f) {
    unsigned u = __builtin_bit_cast(unsigned, f);
    u += 0x7FFFu + ((u >> 16) & 1u);          // round-to-nearest-even
    return (unsigned short)(u >> 16);
}

__global__ void zero_f32(float* p, int n) {
    int i = blockIdx.x * blockDim.x + threadIdx.x;
    if (i < n) p[i] = 0.0f;
}

// per-code: squared norm + bf16 copy of the codebook
__global__ void prep_kernel(const float* __restrict__ cb,
                            float* __restrict__ e2,
                            unsigned short* __restrict__ cb16) {
    int k = blockIdx.x * blockDim.x + threadIdx.x;
    if (k >= KCODES) return;
    float s = 0.0f;
    #pragma unroll 4
    for (int c = 0; c < CDIM; ++c) {
        float v = cb[(size_t)k * CDIM + c];
        s += v * v;
        cb16[(size_t)k * CDIM + c] = f2bf(v);
    }
    e2[k] = s;
}

// one wave per 16-token tile: bf16 WMMA distance ranking + argmin,
// then z_q scatter, loss, EMA statistic accumulation.
__global__ __launch_bounds__(128) void vq_main(
        const float* __restrict__ z,
        const float* __restrict__ cb,
        const float* __restrict__ e2,
        const unsigned short* __restrict__ cb16,
        float* __restrict__ ws_cs,
        float* __restrict__ ws_es,
        float* __restrict__ ws_loss,
        float* __restrict__ out) {
    __shared__ int codesLDS[4][16];

    const int lane = threadIdx.x & 31;
    const int wid  = threadIdx.x >> 5;
    const int tile = blockIdx.x * 4 + wid;     // 0 .. 8191
    const int n0   = tile * 16;
    const int b    = n0 >> 12;                 // 4096 tokens per batch row
    const int t0   = n0 & 4095;
    const int M    = lane & 15;                // token row within tile
    const int hi   = lane >> 4;                // half-group select

    const float* zb = z + (size_t)b * (CDIM * TLEN) + t0;

    // ---- A fragments: 16 tokens x 128 channels, f32 -> bf16 (4 K-chunks of 32)
    // A k order used here (contiguous, B matches): lanes 0-15 row M, elems 0-15 -> K 0..15;
    // lanes 16-31 row M, K 16..31.
    v16bf afr[4];
    #pragma unroll
    for (int kc = 0; kc < 4; ++kc) {
        v16u av;
        #pragma unroll
        for (int e = 0; e < 8; ++e) {
            int c0 = kc * 32 + hi * 16 + e;         // K lo half
            int c1 = kc * 32 + hi * 16 + 8 + e;     // K hi half
            av[e]     = f2bf(zb[(size_t)c0 * TLEN + M]);
            av[e + 8] = f2bf(zb[(size_t)c1 * TLEN + M]);
        }
        afr[kc] = __builtin_bit_cast(v16bf, av);
    }

    // ---- sweep 32 code tiles of 16; score = e2 - 2*dot; running argmin
    float mins[8];
    int   mini[8];
    #pragma unroll
    for (int v = 0; v < 8; ++v) { mins[v] = 3.4e38f; mini[v] = 0; }

    for (int kt = 0; kt < 32; ++kt) {
        const int code16 = kt * 16 + M;     // N index this lane scores
        v8f acc = {};
        #pragma unroll
        for (int kc = 0; kc < 4; ++kc) {
            // B layout: lanes 0-15 hold K = chunk 0..15 (N = lane), lanes 16-31 K = 16..31
            const v16u* bp = (const v16u*)(cb16 + ((size_t)code16 * CDIM + kc * 32 + hi * 16));
            v16bf bfr = __builtin_bit_cast(v16bf, *bp);
            acc = __builtin_amdgcn_wmma_f32_16x16x32_bf16(
                      false, afr[kc], false, bfr, (short)0, acc, false, false);
        }
        const float e2v = e2[code16];
        #pragma unroll
        for (int v = 0; v < 8; ++v) {
            float s = e2v - 2.0f * acc[v];
            if (s < mins[v]) { mins[v] = s; mini[v] = code16; }
        }
    }

    // ---- argmin across the 16 lanes sharing each row (butterfly stays in half-group)
    #pragma unroll
    for (int v = 0; v < 8; ++v) {
        #pragma unroll
        for (int m = 1; m < 16; m <<= 1) {
            float os = __shfl_xor(mins[v], m, 32);
            int   oi = __shfl_xor(mini[v], m, 32);
            if (os < mins[v] || (os == mins[v] && oi < mini[v])) {
                mins[v] = os; mini[v] = oi;
            }
        }
    }
    // rows 0-7 live in lanes 0-15 (vgpr v), rows 8-15 in lanes 16-31
    if (lane == 0) {
        #pragma unroll
        for (int v = 0; v < 8; ++v) codesLDS[wid][v] = mini[v];
    }
    if (lane == 16) {
        #pragma unroll
        for (int v = 0; v < 8; ++v) codesLDS[wid][8 + v] = mini[v];
    }
    __syncthreads();

    const int mycode = codesLDS[wid][M];

    if (lane < 16) {
        int ctok = codesLDS[wid][lane];
        out[O_CODES + n0 + lane] = (float)ctok;
        atomicAdd(&ws_cs[ctok], 1.0f);
    }

    // ---- z_q write ([B,C,T] layout, 16-float coalesced runs), loss, embed_sum
    const float* qrow = cb + (size_t)mycode * CDIM;
    const size_t zofs = (size_t)b * (CDIM * TLEN) + t0 + M;
    float lsum = 0.0f;
    #pragma unroll 4
    for (int it = 0; it < 64; ++it) {
        int cc = it * 2 + hi;
        size_t a = zofs + (size_t)cc * TLEN;
        float zv = z[a];
        float q  = qrow[cc];
        out[O_ZQ + a] = q;                    // z_q_st == z_q numerically
        float d = zv - q;
        lsum += d * d;
        atomicAdd(&ws_es[(size_t)mycode * CDIM + cc], zv);
    }
    #pragma unroll
    for (int off = 16; off; off >>= 1) lsum += __shfl_down(lsum, off, 32);
    if (lane == 0) atomicAdd(ws_loss, lsum);
}

// EMA update + Laplace smoothing + new codebook + loss scale (one block, 512 threads)
__global__ __launch_bounds__(512) void finalize_kernel(
        const float* __restrict__ ecs,
        const float* __restrict__ eemb,
        const float* __restrict__ ws_cs,
        const float* __restrict__ ws_es,
        const float* __restrict__ ws_loss,
        float* __restrict__ out) {
    __shared__ float red[512];
    const int k = threadIdx.x;

    float necs = ecs[k] * DECAYF + ws_cs[k] * ONEMD;
    out[O_ECS + k] = necs;
    red[k] = necs;
    __syncthreads();
    #pragma unroll
    for (int s = 256; s; s >>= 1) {
        if (k < s) red[k] += red[k + s];
        __syncthreads();
    }
    const float n   = red[0];
    const float csm = (necs + EPSF) / (n + (float)KCODES * EPSF) * n;
    const float inv = 1.0f / csm;

    #pragma unroll 4
    for (int c = 0; c < CDIM; ++c) {
        size_t i = (size_t)k * CDIM + c;
        float nee = eemb[i] * DECAYF + ws_es[i] * ONEMD;
        out[O_EEMB + i] = nee;
        out[O_CB + i]   = nee * inv;
    }
    if (k == 0) out[O_LOSS] = ws_loss[0] * (0.25f / 16777216.0f);
}

extern "C" void kernel_launch(void* const* d_in, const int* in_sizes, int n_in,
                              void* d_out, int out_size, void* d_ws, size_t ws_size,
                              hipStream_t stream) {
    const float* z    = (const float*)d_in[0];
    const float* cb   = (const float*)d_in[1];
    const float* ecs  = (const float*)d_in[2];
    const float* eemb = (const float*)d_in[3];
    float* out = (float*)d_out;

    char* ws = (char*)d_ws;
    float*          ws_e2   = (float*)(ws + W_E2);
    unsigned short* ws_cb16 = (unsigned short*)(ws + W_CB16);
    float*          ws_cs   = (float*)(ws + W_CS);
    float*          ws_es   = (float*)(ws + W_ES);
    float*          ws_loss = (float*)(ws + W_LOSS);

    // 1) zero accumulators (cs | es | loss are contiguous)
    zero_f32<<<(W_ZERON + 255) / 256, 256, 0, stream>>>(ws_cs, W_ZERON);
    // 2) codebook norms + bf16 copy
    prep_kernel<<<(KCODES + 127) / 128, 128, 0, stream>>>(cb, ws_e2, ws_cb16);
    // 3) main WMMA quantize pass: 8192 tiles, 4 waves/block
    vq_main<<<2048, 128, 0, stream>>>(z, cb, ws_e2, ws_cb16,
                                      ws_cs, ws_es, ws_loss, out);
    // 4) EMA / smoothing / codebook / loss
    finalize_kernel<<<1, 512, 0, stream>>>(ecs, eemb, ws_cs, ws_es, ws_loss, out);
}
